// MultiHeadAttention_19670950216123
// MI455X (gfx1250) — matrix operand order repaired and checked by reading, compiled, not run on hardware
//
#include <hip/hip_runtime.h>

// ---------------- problem constants ----------------
constexpr int EDIM = 512;
constexpr int SLEN = 2048;
constexpr int NB   = 4;
constexpr int NH   = 8;
constexpr int DH   = 64;   // head dim
constexpr int N3E  = 3 * EDIM;

// ---------------- WMMA types / helpers ----------------
typedef __bf16          v16bf __attribute__((ext_vector_type(16)));
typedef unsigned short  v8us  __attribute__((ext_vector_type(8)));
typedef unsigned short  v16us __attribute__((ext_vector_type(16)));
typedef float           v8f   __attribute__((ext_vector_type(8)));
typedef int             v4i   __attribute__((ext_vector_type(4)));

#if defined(__AMDGCN__) && __has_builtin(__builtin_amdgcn_global_load_async_to_lds_b128)
#define HAS_ASYNC 1
#else
#define HAS_ASYNC 0
#endif

__device__ __forceinline__ unsigned short f2bf(float f) {
    unsigned int u = __float_as_uint(f);
    u += 0x7fffu + ((u >> 16) & 1u);          // round-to-nearest-even
    return (unsigned short)(u >> 16);
}

// copy 16 bytes global -> LDS; async path uses GLOBAL_LOAD_ASYNC_TO_LDS_B128
__device__ __forceinline__ void g2l_b128(const unsigned short* g, unsigned short* l) {
#if HAS_ASYNC
    typedef __attribute__((address_space(1))) v4i* gv4i_p;   // global
    typedef __attribute__((address_space(3))) v4i* lv4i_p;   // LDS
    __builtin_amdgcn_global_load_async_to_lds_b128(
        (gv4i_p)const_cast<unsigned short*>(g), (lv4i_p)l, 0, 0);
#else
    *(v8us*)l = *(const v8us*)g;
#endif
}

__device__ __forceinline__ void async_wait_all() {
#if HAS_ASYNC
    asm volatile("s_wait_asynccnt 0x0" ::: "memory");
#endif
}

__device__ __forceinline__ v16bf frag_from(v8us lo, v8us hi) {
    v16us c;
#pragma unroll
    for (int i = 0; i < 8; ++i) { c[i] = lo[i]; c[i + 8] = hi[i]; }
    return __builtin_bit_cast(v16bf, c);
}

// A fragment (16x32 bf16): p = row base within current 32-wide k window, hl = lane>>4
// lane L holds row L%16; lanes 0-15: K=0..15, lanes 16-31: K=16..31 (consecutive)
__device__ __forceinline__ v16bf load_a(const unsigned short* p, int hl) {
    v8us lo = *(const v8us*)(p + hl * 16);
    v8us hi = *(const v8us*)(p + hl * 16 + 8);
    return frag_from(lo, hi);
}

// B fragment (32x16 bf16) from an [N][K]-major tile: p = &tile[n*ld + ks + hl*16]
// lane L holds column L%16; lanes 0-15: K=0..15, lanes 16-31: K=16..31 (consecutive)
__device__ __forceinline__ v16bf load_b(const unsigned short* p) {
    v8us lo = *(const v8us*)p;
    v8us hi = *(const v8us*)(p + 8);
    return frag_from(lo, hi);
}

__device__ __forceinline__ v8f wmma_bf16(v16bf a, v16bf b, v8f c) {
    return __builtin_amdgcn_wmma_f32_16x16x32_bf16(false, a, false, b, (short)0, c, false, false);
}

// ---------------- kernel 1: weight fp32 -> bf16 ----------------
__global__ __launch_bounds__(256) void convert_weights(
    const float* __restrict__ wq, const float* __restrict__ wo,
    unsigned short* __restrict__ wq_bf, unsigned short* __restrict__ wo_bf)
{
    const int NQ = N3E * EDIM;          // 786432
    const int NO = EDIM * EDIM;         // 262144
    int i4 = (blockIdx.x * blockDim.x + threadIdx.x) * 4;
    if (i4 < NQ) {
        float4 v = *(const float4*)(wq + i4);
        wq_bf[i4 + 0] = f2bf(v.x); wq_bf[i4 + 1] = f2bf(v.y);
        wq_bf[i4 + 2] = f2bf(v.z); wq_bf[i4 + 3] = f2bf(v.w);
    } else {
        int j = i4 - NQ;
        if (j < NO) {
            float4 v = *(const float4*)(wo + j);
            wo_bf[j + 0] = f2bf(v.x); wo_bf[j + 1] = f2bf(v.y);
            wo_bf[j + 2] = f2bf(v.z); wo_bf[j + 3] = f2bf(v.w);
        }
    }
}

// ---------------- kernel 2: QKV projection GEMM ----------------
// C[M,N3E] = x[M,E] @ Wqkv^T + b ; stores Q,K,V bf16 in [B,H,S,D]
// block tile 128x128, 8 waves: wave = 32M x 64N (2x4 wmma tiles)
__global__ __launch_bounds__(256) void qkv_gemm(
    const float* __restrict__ x, const unsigned short* __restrict__ wq_bf,
    const float* __restrict__ b_qkv,
    unsigned short* __restrict__ Q, unsigned short* __restrict__ Kd,
    unsigned short* __restrict__ V)
{
    constexpr int LDT = 40;   // 32 + 8 pad (ushorts), keeps 16B alignment
    __shared__ unsigned short sA[128 * LDT];
    __shared__ unsigned short sB[128 * LDT];

    const int t = threadIdx.x;
    const int w = t >> 5, lane = t & 31, hl = lane >> 4, l16 = lane & 15;
    const int m0 = blockIdx.y * 128, n0 = blockIdx.x * 128;
    const int wm = (w >> 1) * 32, wn = (w & 1) * 64;

    v8f acc[2][4];
#pragma unroll
    for (int mi = 0; mi < 2; ++mi)
#pragma unroll
        for (int ni = 0; ni < 4; ++ni) { v8f z = {}; acc[mi][ni] = z; }

    const int srow = t >> 1, sseg = (t & 1) * 16;

    for (int k0 = 0; k0 < EDIM; k0 += 32) {
        // stage B (bf16 W, [N][K]-major already) -- async global->LDS
        {
            const unsigned short* wp = wq_bf + (size_t)(n0 + srow) * EDIM + k0 + sseg;
            g2l_b128(wp,     sB + srow * LDT + sseg);
            g2l_b128(wp + 8, sB + srow * LDT + sseg + 8);
        }
        // stage A (fp32 -> bf16 conversion, VGPR path)
        {
            const float* xp = x + (size_t)(m0 + srow) * EDIM + k0 + sseg;
            unsigned short* ap = sA + srow * LDT + sseg;
#pragma unroll
            for (int i = 0; i < 4; ++i) {
                float4 v = *(const float4*)(xp + i * 4);
                ap[i * 4 + 0] = f2bf(v.x); ap[i * 4 + 1] = f2bf(v.y);
                ap[i * 4 + 2] = f2bf(v.z); ap[i * 4 + 3] = f2bf(v.w);
            }
        }
        async_wait_all();
        __syncthreads();
#pragma unroll
        for (int mi = 0; mi < 2; ++mi) {
            v16bf a = load_a(sA + (wm + mi * 16 + l16) * LDT, hl);
#pragma unroll
            for (int ni = 0; ni < 4; ++ni) {
                v16bf b = load_b(sB + (wn + ni * 16 + l16) * LDT + hl * 16);
                acc[mi][ni] = wmma_bf16(a, b, acc[mi][ni]);
            }
        }
        __syncthreads();
    }

    // epilogue: bias + scatter to Q/K/V [B,H,S,D] bf16
#pragma unroll
    for (int mi = 0; mi < 2; ++mi)
#pragma unroll
        for (int ni = 0; ni < 4; ++ni) {
            int gn = n0 + wn + ni * 16 + l16;
            float bias = b_qkv[gn];
            int which = gn >> 9;          // 0=Q 1=K 2=V
            int e = gn & 511;
            int h = e >> 6, d = e & 63;
            unsigned short* dstp = (which == 0) ? Q : (which == 1) ? Kd : V;
#pragma unroll
            for (int v = 0; v < 8; ++v) {
                int gm = m0 + wm + mi * 16 + v + 8 * hl;
                int b = gm >> 11, s = gm & 2047;
                dstp[((((size_t)b * NH + h) * SLEN + s) << 6) + d] =
                    f2bf(acc[mi][ni][v] + bias);
            }
        }
}

// ---------------- kernel 3: flash attention ----------------
// grid = (S/128, B*H); 8 waves, each owns 16 Q rows; KV blocks of 64
__global__ __launch_bounds__(256) void flash_attn(
    const unsigned short* __restrict__ Qg, const unsigned short* __restrict__ Kg,
    const unsigned short* __restrict__ Vg, unsigned short* __restrict__ Og)
{
    constexpr int LD = 72;   // 64 + 8 pad ushorts
    __shared__ unsigned short sQ[128 * LD];
    __shared__ unsigned short sK[64 * LD];
    __shared__ unsigned short sVt[64 * LD];        // transposed: [d][s]
    __shared__ unsigned short sP[8 * 16 * LD];     // per-wave 16x64 P tile

    const int t = threadIdx.x, w = t >> 5, lane = t & 31, hl = lane >> 4, l16 = lane & 15;
    const int q0 = blockIdx.x * 128;
    const int bh = blockIdx.y;
    const unsigned short* Qb = Qg + (size_t)bh * SLEN * DH;
    const unsigned short* Kb = Kg + (size_t)bh * SLEN * DH;
    const unsigned short* Vb = Vg + (size_t)bh * SLEN * DH;

    // stage Q tile [128][64] -- async global->LDS
    {
        int row = t >> 1, c0 = (t & 1) * 32;
        const unsigned short* src = Qb + (size_t)(q0 + row) * DH + c0;
        unsigned short* dstp = sQ + row * LD + c0;
#pragma unroll
        for (int i = 0; i < 4; ++i)
            g2l_b128(src + i * 8, dstp + i * 8);
    }

    float mrow[8], lrow[8];
    v8f o_acc[4];
#pragma unroll
    for (int v = 0; v < 8; ++v) { mrow[v] = -3.0e38f; lrow[v] = 0.0f; }
#pragma unroll
    for (int ti = 0; ti < 4; ++ti) { v8f z = {}; o_acc[ti] = z; }

    const float scale = 0.125f;   // 1/sqrt(64)
    const int krow = t >> 2, kc0 = (t & 3) * 16;

    for (int j = 0; j < SLEN / 64; ++j) {
        __syncthreads();   // previous iteration fully consumed sK/sVt
        // stage K [64][64] async; V transposed [d][s] via VGPRs
        {
            const unsigned short* ks = Kb + (size_t)(j * 64 + krow) * DH + kc0;
            g2l_b128(ks,     sK + krow * LD + kc0);
            g2l_b128(ks + 8, sK + krow * LD + kc0 + 8);
            const unsigned short* vs = Vb + (size_t)(j * 64 + krow) * DH + kc0;
            v8us a0 = *(const v8us*)vs, a1 = *(const v8us*)(vs + 8);
#pragma unroll
            for (int i = 0; i < 8; ++i) {
                sVt[(kc0 + i) * LD + krow]     = a0[i];
                sVt[(kc0 + 8 + i) * LD + krow] = a1[i];
            }
        }
        async_wait_all();
        __syncthreads();

        // prefetch next K/V block into GL2 while we compute on this one
        if (j + 1 < SLEN / 64) {
            __builtin_prefetch(Kb + (size_t)((j + 1) * 64 + krow) * DH + kc0, 0, 3);
            __builtin_prefetch(Vb + (size_t)((j + 1) * 64 + krow) * DH + kc0, 0, 3);
        }

        // S = Q K^T  (16x64 per wave)
        v8f sacc[4];
#pragma unroll
        for (int ti = 0; ti < 4; ++ti) { v8f z = {}; sacc[ti] = z; }
#pragma unroll
        for (int ks2 = 0; ks2 < 64; ks2 += 32) {
            v16bf a = load_a(sQ + (w * 16 + l16) * LD + ks2, hl);
#pragma unroll
            for (int ti = 0; ti < 4; ++ti) {
                v16bf b = load_b(sK + (ti * 16 + l16) * LD + ks2 + hl * 16);
                sacc[ti] = wmma_bf16(a, b, sacc[ti]);
            }
        }
#pragma unroll
        for (int ti = 0; ti < 4; ++ti)
#pragma unroll
            for (int v = 0; v < 8; ++v) sacc[ti][v] *= scale;

        // online softmax: rows v+8*hl live across the 16 lanes of this half
        float mnew[8], alpha[8];
#pragma unroll
        for (int v = 0; v < 8; ++v) {
            float mx = fmaxf(fmaxf(sacc[0][v], sacc[1][v]),
                             fmaxf(sacc[2][v], sacc[3][v]));
#pragma unroll
            for (int sh = 1; sh <= 8; sh <<= 1)
                mx = fmaxf(mx, __shfl_xor(mx, sh, 32));
            mnew[v]  = fmaxf(mrow[v], mx);
            alpha[v] = __expf(mrow[v] - mnew[v]);
            mrow[v]  = mnew[v];
        }
#pragma unroll
        for (int ti = 0; ti < 4; ++ti)
#pragma unroll
            for (int v = 0; v < 8; ++v)
                sacc[ti][v] = __expf(sacc[ti][v] - mnew[v]);
#pragma unroll
        for (int v = 0; v < 8; ++v) {
            float sm = sacc[0][v] + sacc[1][v] + sacc[2][v] + sacc[3][v];
#pragma unroll
            for (int sh = 1; sh <= 8; sh <<= 1)
                sm += __shfl_xor(sm, sh, 32);
            lrow[v] = lrow[v] * alpha[v] + sm;
        }
#pragma unroll
        for (int ti = 0; ti < 4; ++ti)
#pragma unroll
            for (int v = 0; v < 8; ++v) o_acc[ti][v] *= alpha[v];

        // P -> LDS (per-wave region), then re-read as A fragments
#pragma unroll
        for (int ti = 0; ti < 4; ++ti)
#pragma unroll
            for (int v = 0; v < 8; ++v)
                sP[(w * 16 + v + 8 * hl) * LD + ti * 16 + l16] = f2bf(sacc[ti][v]);
        asm volatile("s_wait_dscnt 0" ::: "memory");

        // O += P @ V
#pragma unroll
        for (int ks2 = 0; ks2 < 64; ks2 += 32) {
            v16bf a = load_a(sP + (w * 16 + l16) * LD + ks2, hl);
#pragma unroll
            for (int ti = 0; ti < 4; ++ti) {
                v16bf b = load_b(sVt + (ti * 16 + l16) * LD + ks2 + hl * 16);
                o_acc[ti] = wmma_bf16(a, b, o_acc[ti]);
            }
        }
    }

    // normalize + store O as bf16 in [B,S,E] (E = h*64 + d)
    const int h = bh & (NH - 1), b = bh >> 3;
#pragma unroll
    for (int ti = 0; ti < 4; ++ti)
#pragma unroll
        for (int v = 0; v < 8; ++v) {
            int srow = q0 + w * 16 + v + 8 * hl;
            int e = h * DH + ti * 16 + l16;
            Og[((size_t)b * SLEN + srow) * EDIM + e] = f2bf(o_acc[ti][v] / lrow[v]);
        }
}

// ---------------- kernel 4: output projection ----------------
__global__ __launch_bounds__(256) void out_gemm(
    const unsigned short* __restrict__ A, const unsigned short* __restrict__ wo_bf,
    const float* __restrict__ b_out, float* __restrict__ out)
{
    constexpr int LDT = 40;
    __shared__ unsigned short sA[128 * LDT];
    __shared__ unsigned short sB[128 * LDT];

    const int t = threadIdx.x;
    const int w = t >> 5, lane = t & 31, hl = lane >> 4, l16 = lane & 15;
    const int m0 = blockIdx.y * 128, n0 = blockIdx.x * 128;
    const int wm = (w >> 1) * 32, wn = (w & 1) * 64;

    v8f acc[2][4];
#pragma unroll
    for (int mi = 0; mi < 2; ++mi)
#pragma unroll
        for (int ni = 0; ni < 4; ++ni) { v8f z = {}; acc[mi][ni] = z; }

    const int srow = t >> 1, sseg = (t & 1) * 16;

    for (int k0 = 0; k0 < EDIM; k0 += 32) {
        // both tiles are bf16: pure async global->LDS staging
        {
            const unsigned short* ap = A + (size_t)(m0 + srow) * EDIM + k0 + sseg;
            g2l_b128(ap,     sA + srow * LDT + sseg);
            g2l_b128(ap + 8, sA + srow * LDT + sseg + 8);
            const unsigned short* wp = wo_bf + (size_t)(n0 + srow) * EDIM + k0 + sseg;
            g2l_b128(wp,     sB + srow * LDT + sseg);
            g2l_b128(wp + 8, sB + srow * LDT + sseg + 8);
        }
        async_wait_all();
        __syncthreads();
#pragma unroll
        for (int mi = 0; mi < 2; ++mi) {
            v16bf a = load_a(sA + (wm + mi * 16 + l16) * LDT, hl);
#pragma unroll
            for (int ni = 0; ni < 4; ++ni) {
                v16bf b = load_b(sB + (wn + ni * 16 + l16) * LDT + hl * 16);
                acc[mi][ni] = wmma_bf16(a, b, acc[mi][ni]);
            }
        }
        __syncthreads();
    }

#pragma unroll
    for (int mi = 0; mi < 2; ++mi)
#pragma unroll
        for (int ni = 0; ni < 4; ++ni) {
            int gn = n0 + wn + ni * 16 + l16;
            float bias = b_out[gn];
#pragma unroll
            for (int v = 0; v < 8; ++v) {
                int gm = m0 + wm + mi * 16 + v + 8 * hl;
                out[(size_t)gm * EDIM + gn] = acc[mi][ni][v] + bias;
            }
        }
}

// ---------------- launcher ----------------
extern "C" void kernel_launch(void* const* d_in, const int* in_sizes, int n_in,
                              void* d_out, int out_size, void* d_ws, size_t ws_size,
                              hipStream_t stream) {
    (void)in_sizes; (void)n_in; (void)out_size; (void)ws_size;
    const float* x     = (const float*)d_in[0];
    const float* w_qkv = (const float*)d_in[1];
    const float* b_qkv = (const float*)d_in[2];
    const float* w_out = (const float*)d_in[3];
    const float* b_out = (const float*)d_in[4];
    float* out = (float*)d_out;

    char* ws = (char*)d_ws;
    size_t off = 0;
    auto carve = [&](size_t bytes) {
        char* p = ws + off;
        off = (off + bytes + 255) & ~(size_t)255;
        return p;
    };
    unsigned short* wq_bf = (unsigned short*)carve((size_t)N3E * EDIM * 2);
    unsigned short* wo_bf = (unsigned short*)carve((size_t)EDIM * EDIM * 2);
    const size_t qkv_bytes = (size_t)NB * NH * SLEN * DH * 2;
    unsigned short* Q  = (unsigned short*)carve(qkv_bytes);
    unsigned short* K  = (unsigned short*)carve(qkv_bytes);
    unsigned short* V  = (unsigned short*)carve(qkv_bytes);
    unsigned short* O  = (unsigned short*)carve((size_t)NB * SLEN * EDIM * 2);

    // 1) weights -> bf16
    convert_weights<<<1024, 256, 0, stream>>>(w_qkv, w_out, wq_bf, wo_bf);
    // 2) QKV projection: M=8192, N=1536
    qkv_gemm<<<dim3(N3E / 128, (NB * SLEN) / 128), 256, 0, stream>>>(
        x, wq_bf, b_qkv, Q, K, V);
    // 3) flash attention
    flash_attn<<<dim3(SLEN / 128, NB * NH), 256, 0, stream>>>(Q, K, V, O);
    // 4) output projection: M=8192, N=512
    out_gemm<<<dim3(EDIM / 128, (NB * SLEN) / 128), 256, 0, stream>>>(
        O, wo_bf, b_out, out);
}
